// MultiQueryAttention_31112743092238
// MI455X (gfx1250) — matrix operand order repaired and checked by reading, not compiled, not run on hardware
//
#include <hip/hip_runtime.h>
#include <hip/hip_bf16.h>
#include <math.h>

// ---------------------------------------------------------------------------
// MQA forward for MI455X (gfx1250), all GEMMs via v_wmma_f32_16x16x32_bf16,
// LDS staging via GLOBAL_LOAD_ASYNC_TO_LDS_B128 (ASYNCcnt) when available.
// Shapes: B=4, S=2048, E=2048, H=16, Dh=128.
// ---------------------------------------------------------------------------

#define BB 4
#define SS 2048
#define EE 2048
#define HH 16
#define DH 128

typedef __attribute__((ext_vector_type(16))) __bf16 v16bf;
typedef __attribute__((ext_vector_type(8)))  float  v8f;
typedef __attribute__((ext_vector_type(4)))  int    v4i;

#if defined(__HIP_DEVICE_COMPILE__) && \
    __has_builtin(__builtin_amdgcn_global_load_async_to_lds_b128)
#define HAVE_ASYNC_LDS 1
#else
#define HAVE_ASYNC_LDS 0
#endif

#if HAVE_ASYNC_LDS
typedef __attribute__((address_space(1))) v4i gv4i;  // global
typedef __attribute__((address_space(3))) v4i lv4i;  // LDS
#endif

// 16-byte global -> LDS copy. On gfx1250 with the async builtin this lowers to
// global_load_async_to_lds_b128 (no VGPR data path, tracked by ASYNCcnt).
__device__ __forceinline__ void cp16_g2s(const __bf16* g, __bf16* l) {
#if HAVE_ASYNC_LDS
  __builtin_amdgcn_global_load_async_to_lds_b128((gv4i*)g, (lv4i*)l, 0, 0);
#else
  *(uint4*)l = *(const uint4*)g;
#endif
}

__device__ __forceinline__ void wait_async0() {
#if defined(__HIP_DEVICE_COMPILE__) && __has_builtin(__builtin_amdgcn_s_wait_asynccnt)
  __builtin_amdgcn_s_wait_asynccnt(0);
#elif defined(__HIP_DEVICE_COMPILE__)
  asm volatile("s_wait_asynccnt 0" ::: "memory");
#endif
}

__device__ __forceinline__ v8f zero8() {
  v8f v;
#pragma unroll
  for (int i = 0; i < 8; ++i) v[i] = 0.0f;
  return v;
}

// Load a 16-element bf16 fragment as two 16-byte chunks (LDS -> ds_read_b128,
// global -> global_load_b128).
__device__ __forceinline__ v16bf frag_ld(const __bf16* p0, const __bf16* p1) {
  union { uint4 q[2]; v16bf v; } u;
  u.q[0] = *(const uint4*)p0;
  u.q[1] = *(const uint4*)p1;
  return u.v;
}

__device__ __forceinline__ v8f wmma_bf16(v16bf a, v16bf b, v8f c) {
  // (neg_a, A, neg_b, B, c_mod, C, reuse_a, reuse_b)
  return __builtin_amdgcn_wmma_f32_16x16x32_bf16(false, a, false, b,
                                                 (short)0, c, false, false);
}

// ---------------------------------------------------------------------------
// fp32 -> bf16 conversion
// ---------------------------------------------------------------------------
__global__ __launch_bounds__(256) void cvt_f32_bf16(const float* __restrict__ in,
                                                    __bf16* __restrict__ out, int n) {
  int i = blockIdx.x * blockDim.x + threadIdx.x;
  int stride = gridDim.x * blockDim.x;
  for (; i < n; i += stride) out[i] = (__bf16)in[i];
}

// ---------------------------------------------------------------------------
// C[M,N] = A[M,K] @ W[N,K]^T   (A,W bf16 row-major; C bf16 or f32)
// 128x128 block tile, 8 waves (4 along M x 2 along N), K staged 32 at a time,
// double-buffered in LDS with async global->LDS copies.
// ---------------------------------------------------------------------------
template <bool F32OUT>
__global__ __launch_bounds__(256) void gemm_bf16_wmma(const __bf16* __restrict__ A,
                                                      const __bf16* __restrict__ W,
                                                      void* __restrict__ Cout,
                                                      int M, int N, int K) {
  __shared__ __bf16 As[2][128 * 32];
  __shared__ __bf16 Bs[2][128 * 32];

  const int tid  = threadIdx.x;
  const int lane = tid & 31;
  const int w    = tid >> 5;
  const int lmod = lane & 15;
  const bool hi  = lane >= 16;
  const int wm   = w & 3;   // 4 waves along M (32 rows each)
  const int wn   = w >> 2;  // 2 waves along N (64 cols each)
  const int m0   = blockIdx.y * 128;
  const int n0   = blockIdx.x * 128;

  v8f acc[2][4];
#pragma unroll
  for (int mi = 0; mi < 2; ++mi)
#pragma unroll
    for (int ni = 0; ni < 4; ++ni) acc[mi][ni] = zero8();

  const int nk = K >> 5;

  // Prologue: stage slab 0 into buffer 0.
  for (int i = tid; i < 512; i += 256) {
    const int row = i >> 2;
    const int c   = (i & 3) << 3;
    cp16_g2s(A + (size_t)(m0 + row) * K + c, As[0] + row * 32 + c);
    cp16_g2s(W + (size_t)(n0 + row) * K + c, Bs[0] + row * 32 + c);
  }

  for (int kt = 0; kt < nk; ++kt) {
    const int cur = kt & 1;
    // Drain this wave's async copies of slab kt, then join all waves.
    // This barrier also guarantees nobody is still reading buffer cur^1
    // (it was consumed in iteration kt-1), so we may refill it below.
    wait_async0();
    __syncthreads();

    if (kt + 1 < nk) {
      const int nxt = cur ^ 1;
      for (int i = tid; i < 512; i += 256) {
        const int row = i >> 2;
        const int c   = (i & 3) << 3;
        cp16_g2s(A + (size_t)(m0 + row) * K + (kt + 1) * 32 + c, As[nxt] + row * 32 + c);
        cp16_g2s(W + (size_t)(n0 + row) * K + (kt + 1) * 32 + c, Bs[nxt] + row * 32 + c);
      }
    }

    v16bf af[2];
#pragma unroll
    for (int mi = 0; mi < 2; ++mi) {
      const int m  = wm * 32 + mi * 16 + lmod;
      const int kb = hi ? 8 : 0;  // A layout: lanes 0-15 K{0-7,16-23}, 16-31 K{8-15,24-31}
      af[mi] = frag_ld(As[cur] + m * 32 + kb, As[cur] + m * 32 + kb + 16);
    }
#pragma unroll
    for (int ni = 0; ni < 4; ++ni) {
      const int n  = wn * 64 + ni * 16 + lmod;
      const int kb = hi ? 8 : 0;  // B layout: lanes 0-15 K{0-7,16-23}, 16-31 K{8-15,24-31}
      v16bf bf = frag_ld(Bs[cur] + n * 32 + kb, Bs[cur] + n * 32 + kb + 16);
#pragma unroll
      for (int mi = 0; mi < 2; ++mi) acc[mi][ni] = wmma_bf16(af[mi], bf, acc[mi][ni]);
    }
  }

  // Epilogue: C/D layout = lane(N), vgpr r -> M = r + 8*(lane>=16).
#pragma unroll
  for (int mi = 0; mi < 2; ++mi)
#pragma unroll
    for (int ni = 0; ni < 4; ++ni)
#pragma unroll
      for (int r = 0; r < 8; ++r) {
        const int row = m0 + wm * 32 + mi * 16 + r + (hi ? 8 : 0);
        const int col = n0 + wn * 64 + ni * 16 + lmod;
        const float v = acc[mi][ni][r];
        if (F32OUT)
          ((float*)Cout)[(size_t)row * N + col] = v;
        else
          ((__bf16*)Cout)[(size_t)row * N + col] = (__bf16)v;
      }
}

// ---------------------------------------------------------------------------
// Flash attention, multi-query (one KV head broadcast over 16 Q heads).
// grid = (S/128, H, B), block = 256 (8 waves, wave w owns q-rows w*16..w*16+15)
// ---------------------------------------------------------------------------
#define VSTR 136  // V tile row stride ([dh][key]), 16B-aligned

__global__ __launch_bounds__(256) void mqa_flash(const __bf16* __restrict__ Qb,  // [B*S, E]
                                                 const __bf16* __restrict__ Kb,  // [B*S, Dh]
                                                 const __bf16* __restrict__ Vb,  // [B*S, Dh]
                                                 const int* __restrict__ mask,   // [B, S, S]
                                                 __bf16* __restrict__ Ob) {      // [B*S, E]
  __shared__ __bf16 Ks[128 * 128];   // [key][dh]
  __shared__ __bf16 Vs[128 * VSTR];  // [dh][key]  (transposed)
  __shared__ __bf16 Ps[8 * 16 * 128];  // per-wave P tile, [m][key]

  const int qt = blockIdx.x;
  const int h  = blockIdx.y;
  const int b  = blockIdx.z;

  const int tid  = threadIdx.x;
  const int lane = tid & 31;
  const int w    = tid >> 5;
  const int lmod = lane & 15;
  const bool hi  = lane >= 16;
  const int bS   = b * SS;
  const int qrow_base = qt * 128 + w * 16;
  const float scale = 0.088388347648318447f;  // 1/sqrt(128)

  // Q fragments for this wave's 16 rows (A layout), held in registers.
  v16bf qf[4];
  {
    const __bf16* qrow = Qb + (size_t)(bS + qrow_base + lmod) * EE + h * DH;
#pragma unroll
    for (int kd = 0; kd < 4; ++kd) {
      const int kb = kd * 32 + (hi ? 8 : 0);
      qf[kd] = frag_ld(qrow + kb, qrow + kb + 16);
    }
  }

  float m_i[8], l_i[8];
  v8f oacc[8];
#pragma unroll
  for (int r = 0; r < 8; ++r) { m_i[r] = -__builtin_inff(); l_i[r] = 0.0f; }
#pragma unroll
  for (int da = 0; da < 8; ++da) oacc[da] = zero8();

  for (int kt = 0; kt < SS / 128; ++kt) {
    __syncthreads();
    // Stage K tile [key][dh] via async global->LDS (16B per lane-op).
    for (int i = tid; i < 2048; i += 256) {
      const int row = i >> 4;
      const int c   = (i & 15) << 3;
      cp16_g2s(Kb + (size_t)(bS + kt * 128 + row) * DH + c, Ks + row * 128 + c);
    }
    // Stage V tile transposed to [dh][key]: vector global read, scalar LDS scatter.
    for (int i = tid; i < 2048; i += 256) {
      const int key = i >> 4;
      const int dh0 = (i & 15) << 3;
      union { uint4 q; __bf16 e[8]; } u;
      u.q = *(const uint4*)(Vb + (size_t)(bS + kt * 128 + key) * DH + dh0);
#pragma unroll
      for (int j = 0; j < 8; ++j) Vs[(dh0 + j) * VSTR + key] = u.e[j];
    }
    wait_async0();
    __syncthreads();

    // ---- S = Q @ K^T : 8 n-tiles x 4 k-steps of 32 -------------------------
    v8f sacc[8];
#pragma unroll
    for (int nt = 0; nt < 8; ++nt) {
      v8f s = zero8();
      const int col = nt * 16 + lmod;  // key within tile
      const int kbb = hi ? 8 : 0;
#pragma unroll
      for (int kd = 0; kd < 4; ++kd) {
        v16bf kf = frag_ld(Ks + col * 128 + kd * 32 + kbb,
                           Ks + col * 128 + kd * 32 + kbb + 16);
        s = wmma_bf16(qf[kd], kf, s);
      }
      sacc[nt] = s;
    }

    // ---- scale + mask + row max -------------------------------------------
    float rmax[8];
#pragma unroll
    for (int r = 0; r < 8; ++r) rmax[r] = -__builtin_inff();
#pragma unroll
    for (int nt = 0; nt < 8; ++nt) {
      const int key = kt * 128 + nt * 16 + lmod;
#pragma unroll
      for (int r = 0; r < 8; ++r) {
        const int q = qrow_base + r + (hi ? 8 : 0);
        float lg = sacc[nt][r] * scale;
        if (mask[(size_t)b * SS * SS + (size_t)q * SS + key] == 0) lg = -1e9f;
        sacc[nt][r] = lg;
        rmax[r] = fmaxf(rmax[r], lg);
      }
    }
#pragma unroll
    for (int r = 0; r < 8; ++r) {
      rmax[r] = fmaxf(rmax[r], __shfl_xor(rmax[r], 8, 32));
      rmax[r] = fmaxf(rmax[r], __shfl_xor(rmax[r], 4, 32));
      rmax[r] = fmaxf(rmax[r], __shfl_xor(rmax[r], 2, 32));
      rmax[r] = fmaxf(rmax[r], __shfl_xor(rmax[r], 1, 32));
    }

    float m_new[8], corr[8], rsum[8];
#pragma unroll
    for (int r = 0; r < 8; ++r) {
      m_new[r] = fmaxf(m_i[r], rmax[r]);
      corr[r]  = __expf(m_i[r] - m_new[r]);
      rsum[r]  = 0.0f;
    }

    // ---- P = exp(S - m), store bf16 to wave-private LDS (layout transpose) --
    __bf16* pw = Ps + w * (16 * 128);
#pragma unroll
    for (int nt = 0; nt < 8; ++nt) {
      const int keyl = nt * 16 + lmod;
#pragma unroll
      for (int r = 0; r < 8; ++r) {
        const float p = __expf(sacc[nt][r] - m_new[r]);
        rsum[r] += p;
        pw[(r + (hi ? 8 : 0)) * 128 + keyl] = (__bf16)p;
      }
    }
#pragma unroll
    for (int r = 0; r < 8; ++r) {
      rsum[r] += __shfl_xor(rsum[r], 8, 32);
      rsum[r] += __shfl_xor(rsum[r], 4, 32);
      rsum[r] += __shfl_xor(rsum[r], 2, 32);
      rsum[r] += __shfl_xor(rsum[r], 1, 32);
      l_i[r] = l_i[r] * corr[r] + rsum[r];
      m_i[r] = m_new[r];
    }
#pragma unroll
    for (int da = 0; da < 8; ++da)
#pragma unroll
      for (int r = 0; r < 8; ++r) oacc[da][r] *= corr[r];

    // Wave-private LDS region; DS ops are in-order per wave, just drain DScnt
    // before the cross-lane fragment reads.
    asm volatile("s_wait_dscnt 0" ::: "memory");

    // ---- O += P @ V : 4 k-steps of 32 x 8 dh-tiles -------------------------
#pragma unroll
    for (int kd2 = 0; kd2 < 4; ++kd2) {
      const int kba = kd2 * 32 + (hi ? 8 : 0);
      v16bf pf = frag_ld(pw + lmod * 128 + kba, pw + lmod * 128 + kba + 16);
      const int kbb = kd2 * 32 + (hi ? 8 : 0);
#pragma unroll
      for (int da = 0; da < 8; ++da) {
        const int dh = da * 16 + lmod;
        v16bf vf = frag_ld(Vs + dh * VSTR + kbb, Vs + dh * VSTR + kbb + 16);
        oacc[da] = wmma_bf16(pf, vf, oacc[da]);
      }
    }
  }

  // ---- normalize + write O (bf16) ------------------------------------------
#pragma unroll
  for (int da = 0; da < 8; ++da) {
    const int col = h * DH + da * 16 + lmod;
#pragma unroll
    for (int r = 0; r < 8; ++r) {
      const int row = qrow_base + r + (hi ? 8 : 0);
      const float v = oacc[da][r] / l_i[r];
      Ob[(size_t)(bS + row) * EE + col] = (__bf16)v;
    }
  }
}

// ---------------------------------------------------------------------------
// Host launcher
// ---------------------------------------------------------------------------
extern "C" void kernel_launch(void* const* d_in, const int* in_sizes, int n_in,
                              void* d_out, int out_size, void* d_ws, size_t ws_size,
                              hipStream_t stream) {
  (void)in_sizes; (void)n_in; (void)out_size; (void)ws_size;

  const float* x    = (const float*)d_in[0];  // [B,S,E]
  const int*   mask = (const int*)d_in[1];    // [B,1,S,S]
  const float* Wq   = (const float*)d_in[2];  // [E,E]
  const float* Wk   = (const float*)d_in[3];  // [Dh,E]
  const float* Wv   = (const float*)d_in[4];  // [Dh,E]
  const float* Wo   = (const float*)d_in[5];  // [E,E]
  float* out = (float*)d_out;                 // [B,S,E]

  const int MR = BB * SS;  // 8192 rows

  // Workspace layout (bytes)
  char* ws = (char*)d_ws;
  __bf16* xb  = (__bf16*)(ws);                       // 33 554 432
  __bf16* qb  = (__bf16*)(ws + 33554432ull);         // 33 554 432
  __bf16* ob  = (__bf16*)(ws + 67108864ull);         // 33 554 432
  __bf16* wqb = (__bf16*)(ws + 100663296ull);        //  8 388 608
  __bf16* wob = (__bf16*)(ws + 109051904ull);        //  8 388 608
  __bf16* wkb = (__bf16*)(ws + 117440512ull);        //    524 288
  __bf16* wvb = (__bf16*)(ws + 117964800ull);        //    524 288
  __bf16* kb  = (__bf16*)(ws + 118489088ull);        //  2 097 152
  __bf16* vb  = (__bf16*)(ws + 120586240ull);        //  2 097 152  -> ~117 MB total

  // 1) fp32 -> bf16
  cvt_f32_bf16<<<2048, 256, 0, stream>>>(x,  xb,  MR * EE);
  cvt_f32_bf16<<<1024, 256, 0, stream>>>(Wq, wqb, EE * EE);
  cvt_f32_bf16<<<128,  256, 0, stream>>>(Wk, wkb, DH * EE);
  cvt_f32_bf16<<<128,  256, 0, stream>>>(Wv, wvb, DH * EE);
  cvt_f32_bf16<<<1024, 256, 0, stream>>>(Wo, wob, EE * EE);

  // 2) projections: Q = x Wq^T, K = x Wk^T, V = x Wv^T
  gemm_bf16_wmma<false><<<dim3(EE / 128, MR / 128), 256, 0, stream>>>(xb, wqb, qb, MR, EE, EE);
  gemm_bf16_wmma<false><<<dim3(DH / 128, MR / 128), 256, 0, stream>>>(xb, wkb, kb, MR, DH, EE);
  gemm_bf16_wmma<false><<<dim3(DH / 128, MR / 128), 256, 0, stream>>>(xb, wvb, vb, MR, DH, EE);

  // 3) flash attention
  mqa_flash<<<dim3(SS / 128, HH, BB), 256, 0, stream>>>(qb, kb, vb, mask, ob);

  // 4) out = O Wo^T (f32)
  gemm_bf16_wmma<true><<<dim3(EE / 128, MR / 128), 256, 0, stream>>>(ob, wob, out, MR, EE, EE);
}
